// MambaG2G_6536940224931
// MI455X (gfx1250) — matrix operand order repaired and checked by reading, not compiled, not run on hardware
//
#include <hip/hip_runtime.h>
#include <math.h>
#include <stdint.h>

// ---------------- problem constants ----------------
#define N_NODES 1024
#define LSEQ    8
#define DMODEL  1024
#define DI      2048
#define KCONV   4
#define DTRANK  64
#define NEDGE   32768

// ---------------- WMMA types ----------------
typedef __attribute__((ext_vector_type(16))) __bf16 v16bf;
typedef __attribute__((ext_vector_type(8)))  float  v8f;

// ---------------- GEMM tiling ----------------
#define BM 64
#define BN 64
#define BK 32
#define BKW 16          // BK/2 packed bf16x2 words
#define LDW 17          // padded row stride in dwords (gcd(17,64)=1 -> conflict-free)

enum { EP_NONE = 0, EP_TANH_ELU = 1, EP_ELU_P1 = 2, EP_SOFTPLUS = 3 };

__device__ __forceinline__ float sigmoidf_(float x) { return 1.0f / (1.0f + expf(-x)); }

__device__ __forceinline__ uint16_t bfbits16(__bf16 h) {
  union { __bf16 b; uint16_t u; } c; c.b = h; return c.u;
}

// split two f32 into packed bf16x2 hi word + bf16x2 lo word (hi+lo ~= f32)
__device__ __forceinline__ void split_pack2(float f0, float f1, uint32_t& wh, uint32_t& wl) {
  __bf16 h0 = (__bf16)f0, h1 = (__bf16)f1;
  __bf16 l0 = (__bf16)(f0 - (float)h0), l1 = (__bf16)(f1 - (float)h1);
  wh = (uint32_t)bfbits16(h0) | ((uint32_t)bfbits16(h1) << 16);
  wl = (uint32_t)bfbits16(l0) | ((uint32_t)bfbits16(l1) << 16);
}

// A fragment (16x32, ISA 7.12.2): lane L: m=L&15; word j -> K-pair column
// (j&3) + 8*(j>=4) + 4*(L>=16). Pure dword LDS loads, no conversions.
__device__ __forceinline__ void load_a_frag_w(const uint32_t* Ahi, const uint32_t* Alo,
                                              int mBase, int lane, v16bf& hi, v16bf& lo) {
  const int m  = mBase + (lane & 15);
  const int kw = (lane >> 4) << 2;
  const uint32_t* rh = Ahi + m * LDW;
  const uint32_t* rl = Alo + m * LDW;
  union { v16bf v; uint32_t w[8]; } uh, ul;
#pragma unroll
  for (int j = 0; j < 8; ++j) {
    int c = (j & 3) + ((j >> 2) << 3) + kw;
    uh.w[j] = rh[c];
    ul.w[j] = rl[c];
  }
  hi = uh.v; lo = ul.v;
}

// B fragment (32x16): lane L: n=L&15; word j -> K-pair column (j&3) + 8*(j>=4) + 4*(L>=16).
// B planes stored transposed [n][k/2] so K is contiguous.
__device__ __forceinline__ void load_b_frag_w(const uint32_t* Bhi, const uint32_t* Blo,
                                              int nBase, int lane, v16bf& hi, v16bf& lo) {
  const int n  = nBase + (lane & 15);
  const int kw = (lane >> 4) << 2;
  const uint32_t* rh = Bhi + n * LDW + kw;
  const uint32_t* rl = Blo + n * LDW + kw;
  union { v16bf v; uint32_t w[8]; } uh, ul;
#pragma unroll
  for (int j = 0; j < 8; ++j) {
    uh.w[j] = rh[(j & 3) + ((j >> 2) << 3)];
    ul.w[j] = rl[(j & 3) + ((j >> 2) << 3)];
  }
  hi = uh.v; lo = ul.v;
}

// 3-product split-precision accumulate: Ahi*Bhi + Ahi*Blo + Alo*Bhi
__device__ __forceinline__ v8f wmma3(v8f acc, v16bf ah, v16bf al, v16bf bh, v16bf bl) {
  acc = __builtin_amdgcn_wmma_f32_16x16x32_bf16(false, ah, false, bh, (short)0, acc, false, false);
  acc = __builtin_amdgcn_wmma_f32_16x16x32_bf16(false, ah, false, bl, (short)0, acc, false, false);
  acc = __builtin_amdgcn_wmma_f32_16x16x32_bf16(false, al, false, bh, (short)0, acc, false, false);
  return acc;
}

// stage A tile (BM x BK f32 row-major) -> packed bf16 hi/lo planes [m][k/2]
__device__ __forceinline__ void stage_a(const float* __restrict__ A, int lda,
                                        int m0, int k0, int M, int t,
                                        uint32_t* Ahi, uint32_t* Alo) {
  for (int i = t; i < BM * BK / 4; i += 256) {
    int r = i >> 3, c4 = i & 7;
    float4 v = make_float4(0.f, 0.f, 0.f, 0.f);
    int gr = m0 + r;
    if (gr < M) v = *(const float4*)(A + (size_t)gr * lda + k0 + c4 * 4);
    uint32_t h0, l0, h1, l1;
    split_pack2(v.x, v.y, h0, l0);
    split_pack2(v.z, v.w, h1, l1);
    int base = r * LDW + c4 * 2;
    Ahi[base] = h0; Ahi[base + 1] = h1;
    Alo[base] = l0; Alo[base + 1] = l1;
  }
}

// stage B tile (BK x BN f32 row-major) -> transposed packed planes [n][k/2]
__device__ __forceinline__ void stage_b(const float* __restrict__ B, int ldb,
                                        int k0, int n0, int N, int t,
                                        uint32_t* Bhi, uint32_t* Blo) {
  uint16_t* BhiH = (uint16_t*)Bhi;
  uint16_t* BloH = (uint16_t*)Blo;
  for (int i = t; i < BK * BN / 4; i += 256) {
    int r = i >> 4, c4 = i & 15;          // r = k row, 4 consecutive n
    float4 v = make_float4(0.f, 0.f, 0.f, 0.f);
    int gc = n0 + c4 * 4;
    if (gc < N) v = *(const float4*)(B + (size_t)(k0 + r) * ldb + gc);
    float f[4] = {v.x, v.y, v.z, v.w};
    int nl = c4 * 4;
#pragma unroll
    for (int q = 0; q < 4; ++q) {
      __bf16 h = (__bf16)f[q];
      __bf16 l = (__bf16)(f[q] - (float)h);
      BhiH[(nl + q) * (2 * LDW) + r] = bfbits16(h);
      BloH[(nl + q) * (2 * LDW) + r] = bfbits16(l);
    }
  }
}

// ---------------- generic GEMM: C = ep(A@B + bias + add) ----------------
__global__ __launch_bounds__(256)
void gemm_kernel(const float* __restrict__ A, int lda,
                 const float* __restrict__ B, int ldb,
                 const float* __restrict__ bias,
                 const float* __restrict__ addp,
                 float* __restrict__ C, int ldc,
                 int M, int N, int K, int ep)
{
  __shared__ uint32_t Ahi[BM * LDW];
  __shared__ uint32_t Alo[BM * LDW];
  __shared__ uint32_t Bhi[BN * LDW];
  __shared__ uint32_t Blo[BN * LDW];
  const int t = threadIdx.x, lane = t & 31, wave = t >> 5;
  const int mw = wave >> 1, nw = wave & 1;
  const int m0 = blockIdx.y * BM, n0 = blockIdx.x * BN;
  v8f acc0 = {}; v8f acc1 = {};

  for (int k0 = 0; k0 < K; k0 += BK) {
    stage_a(A, lda, m0, k0, M, t, Ahi, Alo);
    stage_b(B, ldb, k0, n0, N, t, Bhi, Blo);
    __syncthreads();
    v16bf ah, al, bh, bl;
    load_a_frag_w(Ahi, Alo, mw * 16, lane, ah, al);
    load_b_frag_w(Bhi, Blo, nw * 32, lane, bh, bl);
    acc0 = wmma3(acc0, ah, al, bh, bl);
    load_b_frag_w(Bhi, Blo, nw * 32 + 16, lane, bh, bl);
    acc1 = wmma3(acc1, ah, al, bh, bl);
    __syncthreads();
  }

  const int gm_base = m0 + mw * 16 + ((lane >> 4) << 3);
  const int gn_base = n0 + nw * 32 + (lane & 15);
#pragma unroll
  for (int v = 0; v < 8; ++v) {
    int gm = gm_base + v;
    if (gm >= M) continue;
#pragma unroll
    for (int tt = 0; tt < 2; ++tt) {
      int gn = gn_base + tt * 16;
      if (gn >= N) continue;
      float x = (tt == 0) ? acc0[v] : acc1[v];
      if (bias) x += bias[gn];
      if (addp) x += addp[(size_t)gm * ldc + gn];
      if (ep == EP_TANH_ELU)      { float th = tanhf(x); x = th > 0.f ? th : expm1f(th); }
      else if (ep == EP_ELU_P1)   { x = (x > 0.f ? x + 1.0f : expm1f(x) + 1.0f) + 1e-14f; }
      else if (ep == EP_SOFTPLUS) { x = x > 20.f ? x : log1pf(expf(x)); }
      C[(size_t)gm * ldc + gn] = x;
    }
  }
}

// --------- fused gate GEMM: H = relu((A@W1+b1)*sigmoid(A@W2+b2) + (A@W3+b3)) ---------
__global__ __launch_bounds__(256)
void gate_gemm_kernel(const float* __restrict__ A, int lda,
                      const float* __restrict__ B1, const float* __restrict__ B2,
                      const float* __restrict__ B3, int ldb,
                      const float* __restrict__ b1, const float* __restrict__ b2,
                      const float* __restrict__ b3,
                      float* __restrict__ H, int ldc, int M, int N, int K)
{
  __shared__ uint32_t Ahi[BM * LDW];
  __shared__ uint32_t Alo[BM * LDW];
  __shared__ uint32_t Bhi[3][BN * LDW];
  __shared__ uint32_t Blo[3][BN * LDW];
  const int t = threadIdx.x, lane = t & 31, wave = t >> 5;
  const int mw = wave >> 1, nw = wave & 1;
  const int m0 = blockIdx.y * BM, n0 = blockIdx.x * BN;
  v8f vz = {};
  v8f acc[3][2];
#pragma unroll
  for (int w = 0; w < 3; ++w) { acc[w][0] = vz; acc[w][1] = vz; }
  const float* Bp[3] = {B1, B2, B3};

  for (int k0 = 0; k0 < K; k0 += BK) {
    stage_a(A, lda, m0, k0, M, t, Ahi, Alo);
#pragma unroll
    for (int w = 0; w < 3; ++w)
      stage_b(Bp[w], ldb, k0, n0, N, t, Bhi[w], Blo[w]);
    __syncthreads();
    v16bf ah, al, bh, bl;
    load_a_frag_w(Ahi, Alo, mw * 16, lane, ah, al);
#pragma unroll
    for (int w = 0; w < 3; ++w) {
      load_b_frag_w(Bhi[w], Blo[w], nw * 32, lane, bh, bl);
      acc[w][0] = wmma3(acc[w][0], ah, al, bh, bl);
      load_b_frag_w(Bhi[w], Blo[w], nw * 32 + 16, lane, bh, bl);
      acc[w][1] = wmma3(acc[w][1], ah, al, bh, bl);
    }
    __syncthreads();
  }

  const int gm_base = m0 + mw * 16 + ((lane >> 4) << 3);
  const int gn_base = n0 + nw * 32 + (lane & 15);
#pragma unroll
  for (int v = 0; v < 8; ++v) {
    int gm = gm_base + v;
    if (gm >= M) continue;
#pragma unroll
    for (int tt = 0; tt < 2; ++tt) {
      int gn = gn_base + tt * 16;
      if (gn >= N) continue;
      float a1 = acc[0][tt][v] + b1[gn];
      float a2 = acc[1][tt][v] + b2[gn];
      float a3 = acc[2][tt][v] + b3[gn];
      float h = a1 * sigmoidf_(a2) + a3;
      H[(size_t)gm * ldc + gn] = h > 0.f ? h : 0.f;
    }
  }
}

// ----- mean over L + transpose: out[d*R + r] = mean_l H[(r*Lr+l)*D + d] -----
__global__ __launch_bounds__(256)
void meanT_kernel(const float* __restrict__ H, float* __restrict__ out,
                  int R, int Lr, int D)
{
  __shared__ float s[32][33];
  int r0 = blockIdx.x * 32, d0 = blockIdx.y * 32;
  int tx = threadIdx.x, ty = threadIdx.y;
  float inv = 1.0f / (float)Lr;
#pragma unroll
  for (int q = 0; q < 4; ++q) {
    int rl = ty + q * 8;
    int r = r0 + rl;
    float acc = 0.f;
    for (int l = 0; l < Lr; ++l)
      acc += H[(size_t)(r * Lr + l) * D + d0 + tx];
    s[rl][tx] = acc * inv;
  }
  __syncthreads();
#pragma unroll
  for (int q = 0; q < 4; ++q) {
    int dl = ty + q * 8;
    out[(size_t)(d0 + dl) * R + r0 + tx] = s[tx][dl];
  }
}

// ----- degree / dinv (integer atomics -> deterministic) -----
__global__ void fill_deg_kernel(int* deg, int n) {
  int i = blockIdx.x * blockDim.x + threadIdx.x;
  if (i < n) deg[i] = 1;   // self-loop
}
__global__ void count_deg_kernel(const int* __restrict__ ei, int* __restrict__ deg, int E) {
  int e = blockIdx.x * blockDim.x + threadIdx.x;
  if (e < E) atomicAdd(&deg[ei[E + e]], 1);
}
__global__ void dinv_kernel(const int* __restrict__ deg, float* __restrict__ dinv, int n) {
  int i = blockIdx.x * blockDim.x + threadIdx.x;
  if (i < n) dinv[i] = rsqrtf((float)deg[i]);
}

// ----- GCN aggregation: one block per destination node, fixed edge order (deterministic) -----
__global__ __launch_bounds__(256)
void gcn_agg_kernel(const float* __restrict__ XW, const int* __restrict__ ei,
                    const float* __restrict__ dinv, const float* __restrict__ b,
                    float* __restrict__ out, int E, int D)
{
  __shared__ int srow[256];
  __shared__ int scol[256];
  const int c = blockIdx.x;
  const int t = threadIdx.x;
  const float dc = dinv[c];
  float acc[4];
#pragma unroll
  for (int q = 0; q < 4; ++q)
    acc[q] = dc * dc * XW[(size_t)c * D + t + 256 * q];  // self loop
  const int* row = ei;
  const int* col = ei + E;
  for (int e0 = 0; e0 < E; e0 += 256) {
    __syncthreads();
    srow[t] = row[e0 + t];
    scol[t] = col[e0 + t];
    __syncthreads();
    for (int k = 0; k < 256; ++k) {
      if (scol[k] == c) {
        int r = srow[k];
        float nm = dinv[r] * dc;
#pragma unroll
        for (int q = 0; q < 4; ++q)
          acc[q] += nm * XW[(size_t)r * D + t + 256 * q];
      }
    }
  }
#pragma unroll
  for (int q = 0; q < 4; ++q) {
    int j = t + 256 * q;
    float v = acc[q] + b[j];
    out[(size_t)c * D + j] = v > 0.f ? v : 0.f;
  }
}

// ----- RMSNorm per row -----
__global__ __launch_bounds__(256)
void rms_kernel(const float* __restrict__ X, const float* __restrict__ w,
                float* __restrict__ Y, int D)
{
  const int b = blockIdx.x, t = threadIdx.x;
  __shared__ float red[256];
  float ss = 0.f;
  for (int j = t; j < D; j += 256) { float v = X[(size_t)b * D + j]; ss += v * v; }
  red[t] = ss;
  __syncthreads();
  for (int s = 128; s > 0; s >>= 1) {
    if (t < s) red[t] += red[t + s];
    __syncthreads();
  }
  const float sc = rsqrtf(red[0] / (float)D + 1e-5f);
  for (int j = t; j < D; j += 256) Y[(size_t)b * D + j] = X[(size_t)b * D + j] * sc * w[j];
}

// ----- depthwise conv (Lt==1 -> only last tap) + SiLU -----
__global__ __launch_bounds__(256)
void conv_silu_kernel(const float* __restrict__ XZ, const float* __restrict__ conv_w,
                      const float* __restrict__ conv_b, float* __restrict__ xc)
{
  int i = blockIdx.x * 256 + threadIdx.x;
  int b = i / DI, j = i - b * DI;
  float x = conv_b[j] + XZ[(size_t)b * (2 * DI) + j] * conv_w[j * KCONV + (KCONV - 1)];
  xc[i] = x * sigmoidf_(x);
}

// ----- closed-form SSM step (Lt==1): u = (delta*xc*(B.C) + Dp*xc) * silu(z) -----
__global__ __launch_bounds__(256)
void ew_y_kernel(const float* __restrict__ dbc, const float* __restrict__ delta,
                 const float* __restrict__ xc, const float* __restrict__ Dp,
                 const float* __restrict__ XZ, float* __restrict__ u)
{
  const int b = blockIdx.x;
  const int t = threadIdx.x;
  __shared__ float sred[16];
  __shared__ float sbc;
  if (t < 16) sred[t] = dbc[(size_t)b * 96 + 64 + t] * dbc[(size_t)b * 96 + 80 + t];
  __syncthreads();
  if (t == 0) { float s = 0.f; for (int i = 0; i < 16; ++i) s += sred[i]; sbc = s; }
  __syncthreads();
  const float bc = sbc;
#pragma unroll
  for (int q = 0; q < 8; ++q) {
    int j = t + 256 * q;
    float xcv = xc[(size_t)b * DI + j];
    float dl  = delta[(size_t)b * DI + j];
    float y   = dl * xcv * bc + Dp[j] * xcv;
    float zv  = XZ[(size_t)b * (2 * DI) + DI + j];
    u[(size_t)b * DI + j] = y * (zv * sigmoidf_(zv));
  }
}

// ---------------- workspace layout (float offsets, heavily aliased) ----------------
#define OFF_H     0              // 8192x1024 gate output (layer1); also XW / XZ / xc / delta
#define OFF_XW    0              // 1024x1024 (after H dead)
#define OFF_XZ    0              // 1024x4096 (after XW dead)
#define OFF_XC    4194304       // 1024x2048
#define OFF_DELTA 6291456       // 1024x2048
#define OFF_MT    8388608       // 1024x1024 (also Hn after mT dead)
#define OFF_G     9437184       // 1024x1024 gcn out / mamba residual
#define OFF_DBC   10485760      // 1024x96
#define OFF_U     10584064      // 1024x2048
#define OFF_E     12681216      // 1024x1024 layer output
#define OFF_DEG   13729792      // 1024 ints
#define OFF_DINV  13730816      // 1024 floats

static inline dim3 gemm_grid(int M, int N) {
  return dim3((unsigned)((N + BN - 1) / BN), (unsigned)((M + BM - 1) / BM));
}

extern "C" void kernel_launch(void* const* d_in, const int* in_sizes, int n_in,
                              void* d_out, int out_size, void* d_ws, size_t ws_size,
                              hipStream_t stream) {
  (void)in_sizes; (void)n_in; (void)out_size; (void)ws_size;
  const float* input   = (const float*)d_in[0];
  const int*   ei      = (const int*)d_in[1];
  const float* tc_W1   = (const float*)d_in[2];
  const float* tc_b1   = (const float*)d_in[3];
  const float* tc_W2   = (const float*)d_in[4];
  const float* tc_b2   = (const float*)d_in[5];
  const float* tc_W3   = (const float*)d_in[6];
  const float* tc_b3   = (const float*)d_in[7];
  const float* gcn_W   = (const float*)d_in[8];
  const float* gcn_b   = (const float*)d_in[9];
  const float* rms_w   = (const float*)d_in[10];
  const float* in_W    = (const float*)d_in[11];
  const float* conv_w  = (const float*)d_in[12];
  const float* conv_b  = (const float*)d_in[13];
  const float* xproj_W = (const float*)d_in[14];
  const float* dt_W    = (const float*)d_in[15];
  const float* dt_b    = (const float*)d_in[16];
  /* d_in[17] = A_log: provably unused (h0 = 0, single scan step) */
  const float* Dp      = (const float*)d_in[18];
  const float* out_W   = (const float*)d_in[19];
  const float* fc_W    = (const float*)d_in[20];
  const float* fc_b    = (const float*)d_in[21];
  const float* mu_W    = (const float*)d_in[22];
  const float* mu_b    = (const float*)d_in[23];
  const float* sig_W   = (const float*)d_in[24];
  const float* sig_b   = (const float*)d_in[25];

  float* ws    = (float*)d_ws;
  float* H     = ws + OFF_H;
  float* XW    = ws + OFF_XW;
  float* XZ    = ws + OFF_XZ;
  float* xc    = ws + OFF_XC;
  float* delta = ws + OFF_DELTA;
  float* mT    = ws + OFF_MT;
  float* Hn    = ws + OFF_MT;   // alias, disjoint lifetime
  float* g     = ws + OFF_G;
  float* dbc   = ws + OFF_DBC;
  float* u     = ws + OFF_U;
  float* e_buf = ws + OFF_E;
  int*   deg   = (int*)(ws + OFF_DEG);
  float* dinv  = ws + OFF_DINV;

  // degrees once (same edge set both layers)
  fill_deg_kernel<<<4, 256, 0, stream>>>(deg, N_NODES);
  count_deg_kernel<<<NEDGE / 256, 256, 0, stream>>>(ei, deg, NEDGE);
  dinv_kernel<<<4, 256, 0, stream>>>(deg, dinv, N_NODES);

  const float* e_cur = input;
  int M1 = N_NODES * LSEQ;   // rows entering the gate GEMM
  int Lr = LSEQ;

  for (int layer = 0; layer < 2; ++layer) {
    // gated temporal MLP (3 fused GEMMs)
    gate_gemm_kernel<<<gemm_grid(M1, DMODEL), 256, 0, stream>>>(
        e_cur, DMODEL, tc_W1, tc_W2, tc_W3, DMODEL, tc_b1, tc_b2, tc_b3,
        H, DMODEL, M1, DMODEL, DMODEL);
    // mean over L + transpose
    meanT_kernel<<<dim3(N_NODES / 32, DMODEL / 32), dim3(32, 8), 0, stream>>>(
        H, mT, N_NODES, Lr, DMODEL);
    // GCN: XW = mT @ gcn_W, then deterministic edge aggregation + bias + relu
    gemm_kernel<<<gemm_grid(DMODEL, N_NODES), 256, 0, stream>>>(
        mT, N_NODES, gcn_W, N_NODES, nullptr, nullptr, XW, N_NODES,
        DMODEL, N_NODES, N_NODES, EP_NONE);
    gcn_agg_kernel<<<N_NODES, 256, 0, stream>>>(XW, ei, dinv, gcn_b, g, NEDGE, DMODEL);
    // mamba (Lt == 1)
    rms_kernel<<<N_NODES, 256, 0, stream>>>(g, rms_w, Hn, DMODEL);
    gemm_kernel<<<gemm_grid(N_NODES, 2 * DI), 256, 0, stream>>>(
        Hn, DMODEL, in_W, 2 * DI, nullptr, nullptr, XZ, 2 * DI,
        N_NODES, 2 * DI, DMODEL, EP_NONE);
    conv_silu_kernel<<<(N_NODES * DI) / 256, 256, 0, stream>>>(XZ, conv_w, conv_b, xc);
    gemm_kernel<<<gemm_grid(N_NODES, 96), 256, 0, stream>>>(
        xc, DI, xproj_W, 96, nullptr, nullptr, dbc, 96,
        N_NODES, 96, DI, EP_NONE);
    gemm_kernel<<<gemm_grid(N_NODES, DI), 256, 0, stream>>>(
        dbc, 96, dt_W, DI, dt_b, nullptr, delta, DI,
        N_NODES, DI, DTRANK, EP_SOFTPLUS);
    ew_y_kernel<<<N_NODES, 256, 0, stream>>>(dbc, delta, xc, Dp, XZ, u);
    gemm_kernel<<<gemm_grid(N_NODES, DMODEL), 256, 0, stream>>>(
        u, DI, out_W, DMODEL, nullptr, g /* residual */, e_buf, DMODEL,
        N_NODES, DMODEL, DI, EP_NONE);

    e_cur = e_buf; M1 = N_NODES; Lr = 1;
  }

  // head: x = elu(tanh(e@fc+fc_b)); mu = x@mu_W+mu_b; sigma = elu(x@sig_W+sig_b)+1+1e-14
  float* xo = (float*)d_out;
  gemm_kernel<<<gemm_grid(N_NODES, DMODEL), 256, 0, stream>>>(
      e_buf, DMODEL, fc_W, DMODEL, fc_b, nullptr, xo, DMODEL,
      N_NODES, DMODEL, DMODEL, EP_TANH_ELU);
  gemm_kernel<<<gemm_grid(N_NODES, 64), 256, 0, stream>>>(
      xo, DMODEL, mu_W, 64, mu_b, nullptr, xo + 1048576, 64,
      N_NODES, 64, DMODEL, EP_NONE);
  gemm_kernel<<<gemm_grid(N_NODES, 64), 256, 0, stream>>>(
      xo, DMODEL, sig_W, 64, sig_b, nullptr, xo + 1048576 + 65536, 64,
      N_NODES, 64, DMODEL, EP_ELU_P1);
}
